// EvacPolicy_2516850835892
// MI455X (gfx1250) — matrix operand order repaired and checked by reading, not compiled, not run on hardware
//
#include <hip/hip_runtime.h>

typedef __attribute__((ext_vector_type(16))) _Float16 v16h;
typedef __attribute__((ext_vector_type(8)))  float    v8f;

// Branch-free exact-GELU: erf via Abramowitz-Stegun 7.1.26 (max abs err 1.5e-7).
// v_rcp_f32 (1 ulp) + v_exp_f32, no divides, no EXEC divergence.
__device__ __forceinline__ float gelu_exact(float x) {
  const float z = fabsf(x) * 0.7071067811865475f;
  const float t = __builtin_amdgcn_rcpf(__builtin_fmaf(0.3275911f, z, 1.0f));
  const float e = __expf(-z * z);
  float p = 1.061405429f;
  p = p * t - 1.453152027f;
  p = p * t + 1.421413741f;
  p = p * t - 0.284496736f;
  p = p * t + 0.254829592f;
  p = p * t;
  const float erf_abs = 1.0f - p * e;          // erf(|x|/sqrt(2))
  const float cdf = 0.5f * (1.0f + copysignf(erf_abs, x));
  return x * cdf;
}

__device__ __forceinline__ v8f wmma_f32_f16(v16h a, v16h b, v8f c) {
  // D = A(16x32 f16) * B(32x16 f16) + C(16x16 f32)
  return __builtin_amdgcn_wmma_f32_16x16x32_f16(false, a, false, b, (short)0, c,
                                                false, false);
}

// A fragment: rows of a 16 x K f16 matrix in LDS (leading dim ldk), K-chunk kc.
// Contiguous K order (matches packed B): lanes 0-15 hold K[kc*32 + 0..15],
// lanes 16-31 hold K[kc*32 + 16..31]; row = lane & 15.
__device__ __forceinline__ v16h a_frag_lds(const _Float16* h, int ldk, int kc, int lane) {
  const int row   = lane & 15;
  const int khalf = lane >> 4;
  const _Float16* r = h + row * ldk + kc * 32 + khalf * 16;
  v16h a;
#pragma unroll
  for (int j = 0; j < 8; ++j) a[j] = r[j];          // 16B contiguous
#pragma unroll
  for (int j = 0; j < 8; ++j) a[8 + j] = r[8 + j];  // 16B contiguous
  return a;
}

// B fragment from fragment-ordered storage: frag f is 512 halfs, lane's 16 halfs
// contiguous -> two 16B vector loads. Packing rule (matches pack_wfrags):
//   dst[f*512 + lane*16 + j] = W[(kc*32 + (lane>>4)*16 + j) * ncols + nt*16 + (lane&15)]
__device__ __forceinline__ v16h b_frag_packed(const _Float16* base, int frag, int lane) {
  const _Float16* p = base + frag * 512 + lane * 16;
  v16h b;
#pragma unroll
  for (int j = 0; j < 16; ++j) b[j] = p[j];
  return b;
}

__global__ void zero_ws(float* p, int n) {
  int i = blockIdx.x * blockDim.x + threadIdx.x;
  if (i < n) p[i] = 0.0f;
}

// Pack a KxN f32 weight matrix (row-major, ncols = N) into fragment-ordered f16.
// frag = kc * nnt + nt ; total = nkc*nnt*512 elements.
__global__ void pack_wfrags(const float* __restrict__ W, int ncols, int nnt,
                            _Float16* __restrict__ dst, int total) {
  int idx = blockIdx.x * blockDim.x + threadIdx.x;
  if (idx >= total) return;
  const int frag = idx >> 9;
  const int rem  = idx & 511;
  const int lane = rem >> 4;
  const int j    = rem & 15;
  const int kc = frag / nnt, nt = frag - kc * nnt;
  const int n = nt * 16 + (lane & 15);
  const int k = kc * 32 + (lane >> 4) * 16 + j;
  dst[idx] = (_Float16)W[k * ncols + n];
}

// ---------------------------------------------------------------------------
// Phase 1: per-node branch MLPs (layer1 VALU, layer2 WMMA f16) fused with
// sorted-segment mean-pool accumulation (LDS per-graph accumulators).
// Block = 256 threads = 8 waves, 128 nodes/block (16 nodes per wave tile).
// ---------------------------------------------------------------------------
#define GACC 16

__global__ __launch_bounds__(256) void node_branches_pool(
    const float* __restrict__ xp, const float* __restrict__ xh,
    const float* __restrict__ xi, const int* __restrict__ batch, int N,
    const float* __restrict__ pedW1, const float* __restrict__ pedB1,
    const float* __restrict__ pedW2, const float* __restrict__ pedB2,
    const float* __restrict__ hazW1, const float* __restrict__ hazB1,
    const float* __restrict__ hazW2, const float* __restrict__ hazB2,
    const float* __restrict__ infW1, const float* __restrict__ infB1,
    const float* __restrict__ infW2, const float* __restrict__ infB2,
    float* __restrict__ seg_sum, float* __restrict__ counts) {
  __shared__ float    w1s[512];          // ped(2x64) | haz(3x64) | inf(3x64)
  __shared__ float    b1s[192];
  __shared__ float    b2s[192];
  __shared__ _Float16 w2f[3 * 8 * 512];  // layer-2 weights, fragment-ordered f16
  __shared__ _Float16 h1buf[128 * 64];   // layer-1 activations (one branch)
  __shared__ float    xbuf[128 * 8];
  __shared__ int      gbuf[128];
  __shared__ float    gacc[GACC * 192];  // per-graph feature accumulators
  __shared__ float    cacc[GACC];

  const int t = threadIdx.x;
  const int base = blockIdx.x * 128;
  const int numNodes = (N - base < 128) ? (N - base) : 128;

  for (int i = t; i < 128; i += 256) w1s[i] = pedW1[i];
  for (int i = t; i < 192; i += 256) w1s[128 + i] = hazW1[i];
  for (int i = t; i < 192; i += 256) w1s[320 + i] = infW1[i];
  if (t < 64) {
    b1s[t] = pedB1[t]; b1s[64 + t] = hazB1[t]; b1s[128 + t] = infB1[t];
    b2s[t] = pedB2[t]; b2s[64 + t] = hazB2[t]; b2s[128 + t] = infB2[t];
  }
  // Stage W2 in fragment order (frag = kc*4 + nt within a branch).
  for (int i = t; i < 3 * 8 * 512; i += 256) {
    const int br   = i / 4096;          // 8 frags * 512
    const int rem  = i - br * 4096;
    const int frag = rem >> 9;          // 0..7
    const int kc = frag >> 2, nt = frag & 3;
    const int r2 = rem & 511;
    const int lane2 = r2 >> 4, j = r2 & 15;
    const int n = nt * 16 + (lane2 & 15);
    const int k = kc * 32 + (lane2 >> 4) * 16 + j;
    const float* W = (br == 0) ? pedW2 : (br == 1 ? hazW2 : infW2);
    w2f[i] = (_Float16)W[k * 64 + n];
  }
  for (int i = t; i < 128; i += 256) {
    const bool ok = (i < numNodes);
    const int node = base + i;
    xbuf[i * 8 + 0] = ok ? xp[node * 2 + 0] : 0.0f;
    xbuf[i * 8 + 1] = ok ? xp[node * 2 + 1] : 0.0f;
    xbuf[i * 8 + 2] = ok ? xh[node * 3 + 0] : 0.0f;
    xbuf[i * 8 + 3] = ok ? xh[node * 3 + 1] : 0.0f;
    xbuf[i * 8 + 4] = ok ? xh[node * 3 + 2] : 0.0f;
    xbuf[i * 8 + 5] = ok ? xi[node * 3 + 0] : 0.0f;
    xbuf[i * 8 + 6] = ok ? xi[node * 3 + 1] : 0.0f;
    xbuf[i * 8 + 7] = ok ? xi[node * 3 + 2] : 0.0f;
    gbuf[i] = ok ? batch[node] : 0;
  }
  for (int i = t; i < GACC * 192; i += 256) gacc[i] = 0.0f;
  if (t < GACC) cacc[t] = 0.0f;
  __syncthreads();

  // batch is sorted -> block's graphs form a contiguous range.
  const int gmin = gbuf[0];
  const int gmax = gbuf[numNodes - 1];
  const bool useLds = (gmax - gmin) < GACC;

  if (t < numNodes) {
    const int g = gbuf[t];
    if (useLds) atomicAdd(&cacc[g - gmin], 1.0f);
    else        atomicAdd(&counts[g], 1.0f);
  }

  const int wave  = t >> 5;
  const int lane  = t & 31;
  const int nlo   = lane & 15;
  const int khalf = lane >> 4;

  // This lane's 8 accumulator rows cover consecutive nodes n0..n0+7 (sorted
  // batch => usually one graph). Detect the single-graph fast path once.
  const int n0 = wave * 16 + 8 * khalf;
  const int gFirst = gbuf[n0];
  const int gLast  = gbuf[(n0 + 7) < 127 ? (n0 + 7) : 127];
  const bool oneGraph = (n0 + 7 < numNodes) && (gFirst == gLast);

#pragma unroll
  for (int br = 0; br < 3; ++br) {
    const int xoff  = (br == 0) ? 0 : (br == 1 ? 2 : 5);
    const int dk    = (br == 0) ? 2 : 3;
    const int w1off = (br == 0) ? 0 : (br == 1 ? 128 : 320);
    __syncthreads();  // protect h1buf reuse across branches
    // layer 1: h1 = gelu(x @ W1 + b1), 128 nodes x 64 dims
    for (int i = t; i < 128 * 64; i += 256) {
      const int nodeL = i >> 6, dim = i & 63;
      float h = b1s[br * 64 + dim];
      for (int k = 0; k < dk; ++k)
        h += xbuf[nodeL * 8 + xoff + k] * w1s[w1off + k * 64 + dim];
      h1buf[i] = (_Float16)gelu_exact(h);
    }
    __syncthreads();
    // layer 2 via WMMA: (16x64) @ (64x64), one 16-node tile per wave.
    const _Float16* hW = &h1buf[wave * 16 * 64];
    const _Float16* WF = &w2f[br * 8 * 512];
    const v16h a0 = a_frag_lds(hW, 64, 0, lane);  // hoisted across nt
    const v16h a1 = a_frag_lds(hW, 64, 1, lane);
#pragma unroll
    for (int nt = 0; nt < 4; ++nt) {
      const float bias = b2s[br * 64 + nt * 16 + nlo];
      v8f acc;
#pragma unroll
      for (int r = 0; r < 8; ++r) acc[r] = bias;
      acc = wmma_f32_f16(a0, b_frag_packed(WF, 0 * 4 + nt, lane), acc);
      acc = wmma_f32_f16(a1, b_frag_packed(WF, 1 * 4 + nt, lane), acc);
      // GELU + fused segment-sum accumulation (C layout: reg r -> M = r+8*khalf)
      const int feat = br * 64 + nt * 16 + nlo;
      float vals[8];
#pragma unroll
      for (int r = 0; r < 8; ++r) vals[r] = gelu_exact(acc[r]);
      if (oneGraph) {
        // All 8 rows belong to one graph: reduce in registers, 1 atomic.
        float s = 0.0f;
#pragma unroll
        for (int r = 0; r < 8; ++r) s += vals[r];
        if (useLds) atomicAdd(&gacc[(gFirst - gmin) * 192 + feat], s);
        else        atomicAdd(&seg_sum[(size_t)gFirst * 192 + feat], s);
      } else {
#pragma unroll
        for (int r = 0; r < 8; ++r) {
          const int nodeL = n0 + r;
          if (nodeL < numNodes) {
            const int g = gbuf[nodeL];
            if (useLds) atomicAdd(&gacc[(g - gmin) * 192 + feat], vals[r]);
            else        atomicAdd(&seg_sum[(size_t)g * 192 + feat], vals[r]);
          }
        }
      }
    }
  }
  __syncthreads();
  if (useLds) {
    const int R = gmax - gmin + 1;
    for (int i = t; i < R * 192; i += 256) {
      const int q = i / 192, f = i - q * 192;
      atomicAdd(&seg_sum[(size_t)(gmin + q) * 192 + f], gacc[i]);
    }
    if (t < R) atomicAdd(&counts[gmin + t], cacc[t]);
  }
}

// ---------------------------------------------------------------------------
// Phase 2: finalize segment mean, then all graph-head GEMMs with WMMA f16.
// One wave per tile of 16 graphs; B fragments read from fragment-ordered f16
// weights staged in workspace (coalesced 16B vector loads, L2-resident).
// ---------------------------------------------------------------------------
__global__ __launch_bounds__(32) void graph_heads(
    const float* __restrict__ seg_sum, const float* __restrict__ counts,
    const _Float16* __restrict__ fc1P, const float* __restrict__ fc1b,
    const _Float16* __restrict__ fc2P, const float* __restrict__ fc2b,
    const float* __restrict__ shW, const float* __restrict__ shb,
    const float* __restrict__ gdW, const float* __restrict__ gdb,
    const _Float16* __restrict__ c1P, const float* __restrict__ c1b,
    const _Float16* __restrict__ c2P, const float* __restrict__ c2b,
    const float* __restrict__ c3W, const float* __restrict__ c3b,
    float* __restrict__ out, int B) {
  __shared__ _Float16 ge[16 * 192];
  __shared__ _Float16 d1[16 * 256];  // fc1 out; later reused as critic hidden 16x128
  __shared__ _Float16 d2[16 * 128];
  __shared__ _Float16 v2s[16 * 64];

  const int lane  = threadIdx.x;
  const int gbase = blockIdx.x * 16;
  const int nlo   = lane & 15;
  const int khalf = lane >> 4;

  // graph_embed = seg_sum / max(count, 1)
  for (int i = lane; i < 16 * 192; i += 32) {
    const int m = i / 192, f = i - m * 192;
    const int g = gbase + m;
    float val = 0.0f;
    if (g < B) val = seg_sum[(size_t)g * 192 + f] / fmaxf(counts[g], 1.0f);
    ge[i] = (_Float16)val;
  }
  __syncthreads();

  v16h aGe[6];
#pragma unroll
  for (int kc = 0; kc < 6; ++kc) aGe[kc] = a_frag_lds(ge, 192, kc, lane);

  // fc1: 192 -> 256, GELU
#pragma unroll
  for (int nt = 0; nt < 16; ++nt) {
    const float bias = fc1b[nt * 16 + nlo];
    v8f acc;
#pragma unroll
    for (int r = 0; r < 8; ++r) acc[r] = bias;
#pragma unroll
    for (int kc = 0; kc < 6; ++kc)
      acc = wmma_f32_f16(aGe[kc], b_frag_packed(fc1P, kc * 16 + nt, lane), acc);
#pragma unroll
    for (int r = 0; r < 8; ++r)
      d1[(r + 8 * khalf) * 256 + nt * 16 + nlo] = (_Float16)gelu_exact(acc[r]);
  }
  __syncthreads();

  v16h aD1[8];
#pragma unroll
  for (int kc = 0; kc < 8; ++kc) aD1[kc] = a_frag_lds(d1, 256, kc, lane);

  // fc2: 256 -> 128, GELU
#pragma unroll
  for (int nt = 0; nt < 8; ++nt) {
    const float bias = fc2b[nt * 16 + nlo];
    v8f acc;
#pragma unroll
    for (int r = 0; r < 8; ++r) acc[r] = bias;
#pragma unroll
    for (int kc = 0; kc < 8; ++kc)
      acc = wmma_f32_f16(aD1[kc], b_frag_packed(fc2P, kc * 8 + nt, lane), acc);
#pragma unroll
    for (int r = 0; r < 8; ++r)
      d2[(r + 8 * khalf) * 128 + nt * 16 + nlo] = (_Float16)gelu_exact(acc[r]);
  }
  __syncthreads();

  // critic c1: 192 -> 128, GELU (reuse d1 as 16x128)
#pragma unroll
  for (int nt = 0; nt < 8; ++nt) {
    const float bias = c1b[nt * 16 + nlo];
    v8f acc;
#pragma unroll
    for (int r = 0; r < 8; ++r) acc[r] = bias;
#pragma unroll
    for (int kc = 0; kc < 6; ++kc)
      acc = wmma_f32_f16(aGe[kc], b_frag_packed(c1P, kc * 8 + nt, lane), acc);
#pragma unroll
    for (int r = 0; r < 8; ++r)
      d1[(r + 8 * khalf) * 128 + nt * 16 + nlo] = (_Float16)gelu_exact(acc[r]);
  }
  __syncthreads();

  v16h aV1[4];
#pragma unroll
  for (int kc = 0; kc < 4; ++kc) aV1[kc] = a_frag_lds(d1, 128, kc, lane);

  // critic c2: 128 -> 64, GELU
#pragma unroll
  for (int nt = 0; nt < 4; ++nt) {
    const float bias = c2b[nt * 16 + nlo];
    v8f acc;
#pragma unroll
    for (int r = 0; r < 8; ++r) acc[r] = bias;
#pragma unroll
    for (int kc = 0; kc < 4; ++kc)
      acc = wmma_f32_f16(aV1[kc], b_frag_packed(c2P, kc * 4 + nt, lane), acc);
#pragma unroll
    for (int r = 0; r < 8; ++r)
      v2s[(r + 8 * khalf) * 64 + nt * 16 + nlo] = (_Float16)gelu_exact(acc[r]);
  }
  __syncthreads();

  // Final 1-wide heads in VALU (tiny dots, f32)
  if (lane < 16) {
    const int g = gbase + lane;
    if (g < B) {
      float s1 = 0.0f, s2 = 0.0f;
      for (int f = 0; f < 128; ++f) {
        const float d = (float)d2[lane * 128 + f];
        s1 += d * shW[f];
        s2 += d * gdW[f];
      }
      float s3 = 0.0f;
      for (int f = 0; f < 64; ++f) s3 += (float)v2s[lane * 64 + f] * c3W[f];
      out[g]         = s1 + shb[0];   // shelter_logits
      out[B + g]     = s2 + gdb[0];   // guidance_logits
      out[2 * B + g] = s3 + c3b[0];   // value
    }
  }
}

extern "C" void kernel_launch(void* const* d_in, const int* in_sizes, int n_in,
                              void* d_out, int out_size, void* d_ws, size_t ws_size,
                              hipStream_t stream) {
  (void)n_in; (void)ws_size;
  const float* xp    = (const float*)d_in[0];
  const float* xh    = (const float*)d_in[1];
  const float* xi    = (const float*)d_in[2];
  const int*   batch = (const int*)d_in[3];
  // d_in[4] = num_graphs (device scalar, unused: B derived from out_size)
  const float* pedW1 = (const float*)d_in[5];
  const float* pedB1 = (const float*)d_in[6];
  const float* pedW2 = (const float*)d_in[7];
  const float* pedB2 = (const float*)d_in[8];
  const float* hazW1 = (const float*)d_in[9];
  const float* hazB1 = (const float*)d_in[10];
  const float* hazW2 = (const float*)d_in[11];
  const float* hazB2 = (const float*)d_in[12];
  const float* infW1 = (const float*)d_in[13];
  const float* infB1 = (const float*)d_in[14];
  const float* infW2 = (const float*)d_in[15];
  const float* infB2 = (const float*)d_in[16];
  const float* fc1W  = (const float*)d_in[17];
  const float* fc1b  = (const float*)d_in[18];
  const float* fc2W  = (const float*)d_in[19];
  const float* fc2b  = (const float*)d_in[20];
  const float* shW   = (const float*)d_in[21];
  const float* shb   = (const float*)d_in[22];
  const float* gdW   = (const float*)d_in[23];
  const float* gdb   = (const float*)d_in[24];
  const float* c1W   = (const float*)d_in[25];
  const float* c1b   = (const float*)d_in[26];
  const float* c2W   = (const float*)d_in[27];
  const float* c2b   = (const float*)d_in[28];
  const float* c3W   = (const float*)d_in[29];
  const float* c3b   = (const float*)d_in[30];

  const int N = in_sizes[0] / 2;     // x_ped is (N, 2)
  const int B = out_size / 3;        // [shelter | guidance | value]

  // Workspace layout: seg_sum (B*192 f32) | counts (B f32) | packed f16 weights
  float* seg_sum = (float*)d_ws;
  float* counts  = seg_sum + (size_t)B * 192;
  _Float16* fc1P = (_Float16*)(counts + B);      // 6*16 frags = 49152 halfs
  _Float16* fc2P = fc1P + 6 * 16 * 512;          // 8*8  frags = 32768 halfs
  _Float16* c1P  = fc2P + 8 * 8 * 512;           // 6*8  frags = 24576 halfs
  _Float16* c2P  = c1P + 6 * 8 * 512;            // 4*4  frags =  8192 halfs

  const int zn = B * 192 + B;
  zero_ws<<<(zn + 255) / 256, 256, 0, stream>>>(seg_sum, zn);

  // Stage head weights in fragment order (tiny; runs after zeroing on the same stream).
  pack_wfrags<<<(6 * 16 * 512 + 255) / 256, 256, 0, stream>>>(fc1W, 256, 16, fc1P, 6 * 16 * 512);
  pack_wfrags<<<(8 * 8 * 512 + 255) / 256, 256, 0, stream>>>(fc2W, 128, 8, fc2P, 8 * 8 * 512);
  pack_wfrags<<<(6 * 8 * 512 + 255) / 256, 256, 0, stream>>>(c1W, 128, 8, c1P, 6 * 8 * 512);
  pack_wfrags<<<(4 * 4 * 512 + 255) / 256, 256, 0, stream>>>(c2W, 64, 4, c2P, 4 * 4 * 512);

  const int nb1 = (N + 127) / 128;
  node_branches_pool<<<nb1, 256, 0, stream>>>(
      xp, xh, xi, batch, N,
      pedW1, pedB1, pedW2, pedB2,
      hazW1, hazB1, hazW2, hazB2,
      infW1, infB1, infW2, infB2,
      seg_sum, counts);

  graph_heads<<<(B + 15) / 16, 32, 0, stream>>>(
      seg_sum, counts,
      fc1P, fc1b, fc2P, fc2b, shW, shb, gdW, gdb,
      c1P, c1b, c2P, c2b, c3W, c3b,
      (float*)d_out, B);
}
